// SampleQueryMessagePassing_86852828660040
// MI455X (gfx1250) — matrix operand order repaired and checked by reading, not compiled, not run on hardware
//
#include <hip/hip_runtime.h>
#include <hip/hip_bf16.h>
#include <math.h>

typedef __attribute__((ext_vector_type(16))) __bf16 v16bf;
typedef __attribute__((ext_vector_type(8)))  __bf16 bf16x8;
typedef __attribute__((ext_vector_type(4)))  __bf16 bf16x4;
typedef __attribute__((ext_vector_type(8)))  float  v8f;

#define BB   64
#define SS   4096
#define CC   128
#define MID  512
#define TOK  64          // tokens per workgroup (4 M-tiles of 16)
#define XP   264         // X pitch   (256 + 8 bf16 pad)
#define HP   520         // H pitch   (512 + 8 bf16 pad)

// gelu(tanh approx): 0.5*x*(1+tanh(z)) == x*sigmoid(2z); branch-free exp+rcp
__device__ __forceinline__ float gelu_fast(float x) {
    float z = x + 0.044715f * x * x * x;
    return x * __builtin_amdgcn_rcpf(1.0f + __expf(-1.5957691216057308f * z));
}
__device__ __forceinline__ float sigmoid_fast(float x) {
    return __builtin_amdgcn_rcpf(1.0f + __expf(-x));
}

// ---------------------------------------------------------------------------
// Kernel 0: fp32 weights -> bf16, transposed to [N][K]; also zero agg buffer.
// ---------------------------------------------------------------------------
__global__ __launch_bounds__(256) void convert_weights(
    const float* __restrict__ W0, const float* __restrict__ W1,
    const float* __restrict__ Wout,
    __bf16* __restrict__ W0T, __bf16* __restrict__ W1T, __bf16* __restrict__ WoT,
    float* __restrict__ agg)
{
    int i = blockIdx.x * 256 + threadIdx.x;          // grid covers 512*512
    if (i < 512 * 256) {                             // W0: [256][512] -> W0T [512][256]
        int n = i >> 8, k = i & 255;
        W0T[i] = (__bf16)W0[k * 512 + n];
    }
    if (i < 512 * 512) {                             // W1: [512][512] -> W1T [512][512]
        int n = i >> 9, k = i & 511;
        W1T[i] = (__bf16)W1[k * 512 + n];
    }
    if (i < 128 * 512) {                             // Wout: [512][128] -> WoT [128][512]
        int n = i >> 9, k = i & 511;
        WoT[i] = (__bf16)Wout[k * 128 + n];
    }
    if (i < BB * CC) agg[i] = 0.0f;
}

// ---------------------------------------------------------------------------
// Kernel 1: fused X-build + GEMM0(gelu) + GEMM1(gelu) + GEMM2(sigmoid) + sum.
// 256 threads = 8 waves; wave = (m-tile, n-half). 64 tokens per block.
// n-tiles processed in pairs: shared A-fragment, two independent WMMA chains.
// ---------------------------------------------------------------------------
__global__ __launch_bounds__(256) void fused_mlp(
    const float* __restrict__ sf, const float* __restrict__ qs,
    const float* __restrict__ pe,
    const __bf16* __restrict__ W0T, const float* __restrict__ b0,
    const __bf16* __restrict__ W1T, const float* __restrict__ b1,
    const __bf16* __restrict__ WoT, const float* __restrict__ bout,
    float* __restrict__ agg)
{
    __shared__ __bf16 sH0[TOK * HP];                 // 66560 B
    __shared__ __bf16 sXB[TOK * HP];                 // X (pitch XP) then H1 (pitch HP)

    const int tid  = threadIdx.x;
    const int lane = tid & 31;
    const int wid  = tid >> 5;
    const int tok0 = blockIdx.x * TOK;
    const int batch = tok0 >> 12;                    // 4096 tokens per batch

    // ---- Stage 0: X = [sf - (qs+pe), (qs+pe)]  (bf16 in LDS, float4 loads) ----
    for (int idx = tid; idx < TOK * CC / 4; idx += 256) {
        int r  = idx >> 5;                            // 32 float4 slots per row
        int c4 = (idx & 31) * 4;
        int g  = (tok0 + r) * CC + c4;
        float4 sv = *(const float4*)(sf + g);
        float4 qa = *(const float4*)(qs + g);
        float4 pa = *(const float4*)(pe + g);
        float4 qv = make_float4(qa.x + pa.x, qa.y + pa.y, qa.z + pa.z, qa.w + pa.w);
        bf16x4 d, q;
        d[0] = (__bf16)(sv.x - qv.x); d[1] = (__bf16)(sv.y - qv.y);
        d[2] = (__bf16)(sv.z - qv.z); d[3] = (__bf16)(sv.w - qv.w);
        q[0] = (__bf16)qv.x; q[1] = (__bf16)qv.y;
        q[2] = (__bf16)qv.z; q[3] = (__bf16)qv.w;
        *(bf16x4*)(&sXB[r * XP + c4])       = d;
        *(bf16x4*)(&sXB[r * XP + 128 + c4]) = q;
    }
    __syncthreads();

    const int mt = wid & 3, nh = wid >> 2;
    const int mr = mt * 16;
    const int cl = lane & 15, hi = lane >> 4;

    // ---- GEMM0: X[64x256] @ W0[256x512] + b0, GELU -> H0 ----
    for (int nt = nh * 16; nt < nh * 16 + 16; nt += 2) {
        const int n0 = nt * 16 + cl, n1 = n0 + 16;
        float bias0 = b0[n0], bias1 = b0[n1];
        v8f acc0, acc1;
        #pragma unroll
        for (int i = 0; i < 8; ++i) { acc0[i] = bias0; acc1[i] = bias1; }
        #pragma unroll
        for (int kt = 0; kt < 8; ++kt) {
            union { v16bf v; bf16x8 h[2]; } a;
            const __bf16* ap = &sXB[(mr + cl) * XP + kt * 32 + hi * 16];
            a.h[0] = *(const bf16x8*)ap;
            a.h[1] = *(const bf16x8*)(ap + 8);
            v16bf bf0 = *(const v16bf*)(W0T + n0 * 256 + kt * 32 + hi * 16);
            v16bf bf1 = *(const v16bf*)(W0T + n1 * 256 + kt * 32 + hi * 16);
            acc0 = __builtin_amdgcn_wmma_f32_16x16x32_bf16(
                       false, a.v, false, bf0, (short)0, acc0, false, false);
            acc1 = __builtin_amdgcn_wmma_f32_16x16x32_bf16(
                       false, a.v, false, bf1, (short)0, acc1, false, false);
        }
        #pragma unroll
        for (int r = 0; r < 8; ++r) {
            sH0[(mr + r + hi * 8) * HP + n0] = (__bf16)gelu_fast(acc0[r]);
            sH0[(mr + r + hi * 8) * HP + n1] = (__bf16)gelu_fast(acc1[r]);
        }
    }
    __syncthreads();   // H0 ready; X dead (sXB may be overwritten as H1)

    // ---- GEMM1: H0[64x512] @ W1[512x512] + b1, GELU -> H1 (in sXB) ----
    for (int nt = nh * 16; nt < nh * 16 + 16; nt += 2) {
        const int n0 = nt * 16 + cl, n1 = n0 + 16;
        float bias0 = b1[n0], bias1 = b1[n1];
        v8f acc0, acc1;
        #pragma unroll
        for (int i = 0; i < 8; ++i) { acc0[i] = bias0; acc1[i] = bias1; }
        #pragma unroll
        for (int kt = 0; kt < 16; ++kt) {
            union { v16bf v; bf16x8 h[2]; } a;
            const __bf16* ap = &sH0[(mr + cl) * HP + kt * 32 + hi * 16];
            a.h[0] = *(const bf16x8*)ap;
            a.h[1] = *(const bf16x8*)(ap + 8);
            v16bf bf0 = *(const v16bf*)(W1T + n0 * 512 + kt * 32 + hi * 16);
            v16bf bf1 = *(const v16bf*)(W1T + n1 * 512 + kt * 32 + hi * 16);
            acc0 = __builtin_amdgcn_wmma_f32_16x16x32_bf16(
                       false, a.v, false, bf0, (short)0, acc0, false, false);
            acc1 = __builtin_amdgcn_wmma_f32_16x16x32_bf16(
                       false, a.v, false, bf1, (short)0, acc1, false, false);
        }
        #pragma unroll
        for (int r = 0; r < 8; ++r) {
            sXB[(mr + r + hi * 8) * HP + n0] = (__bf16)gelu_fast(acc0[r]);
            sXB[(mr + r + hi * 8) * HP + n1] = (__bf16)gelu_fast(acc1[r]);
        }
    }
    __syncthreads();   // H1 ready

    // ---- GEMM2: H1[64x512] @ Wout[512x128] + bout, sigmoid, row-sum ----
    for (int nt = nh * 4; nt < nh * 4 + 4; nt += 2) {
        const int n0 = nt * 16 + cl, n1 = n0 + 16;
        float bias0 = bout[n0], bias1 = bout[n1];
        v8f acc0, acc1;
        #pragma unroll
        for (int i = 0; i < 8; ++i) { acc0[i] = bias0; acc1[i] = bias1; }
        #pragma unroll
        for (int kt = 0; kt < 16; ++kt) {
            union { v16bf v; bf16x8 h[2]; } a;
            const __bf16* ap = &sXB[(mr + cl) * HP + kt * 32 + hi * 16];
            a.h[0] = *(const bf16x8*)ap;
            a.h[1] = *(const bf16x8*)(ap + 8);
            v16bf bf0 = *(const v16bf*)(WoT + n0 * 512 + kt * 32 + hi * 16);
            v16bf bf1 = *(const v16bf*)(WoT + n1 * 512 + kt * 32 + hi * 16);
            acc0 = __builtin_amdgcn_wmma_f32_16x16x32_bf16(
                       false, a.v, false, bf0, (short)0, acc0, false, false);
            acc1 = __builtin_amdgcn_wmma_f32_16x16x32_bf16(
                       false, a.v, false, bf1, (short)0, acc1, false, false);
        }
        float s0 = 0.0f, s1 = 0.0f;
        #pragma unroll
        for (int r = 0; r < 8; ++r) {
            s0 += sigmoid_fast(acc0[r]);
            s1 += sigmoid_fast(acc1[r]);
        }
        atomicAdd(&agg[batch * CC + n0], s0);
        atomicAdd(&agg[batch * CC + n1], s1);
    }
}

// ---------------------------------------------------------------------------
// Kernel 2: LayerNorm of agg[B,1,C] over C=128. One block per batch.
// ---------------------------------------------------------------------------
__global__ __launch_bounds__(128) void agg_ln(
    const float* __restrict__ agg, const float* __restrict__ mg,
    const float* __restrict__ mb, float* __restrict__ aggn)
{
    __shared__ float red[128];
    int b = blockIdx.x, c = threadIdx.x;
    float x = agg[b * CC + c];

    red[c] = x; __syncthreads();
    for (int s = 64; s > 0; s >>= 1) { if (c < s) red[c] += red[c + s]; __syncthreads(); }
    float mu = red[0] * (1.0f / 128.0f);
    __syncthreads();

    float d = x - mu;
    red[c] = d * d; __syncthreads();
    for (int s = 64; s > 0; s >>= 1) { if (c < s) red[c] += red[c + s]; __syncthreads(); }
    float var = red[0] * (1.0f / 128.0f);

    aggn[b * CC + c] = d * rsqrtf(var + 1e-3f) * mg[c] + mb[c];
}

// ---------------------------------------------------------------------------
// Kernel 3: out = LayerNorm(sf + aggn) per row of 128 channels.
// 8 waves per block, one wave per row; float4 per lane; shfl reductions.
// ---------------------------------------------------------------------------
__global__ __launch_bounds__(256) void out_ln(
    const float* __restrict__ sf, const float* __restrict__ aggn,
    const float* __restrict__ og, const float* __restrict__ ob,
    float* __restrict__ out)
{
    int lane = threadIdx.x & 31, w = threadIdx.x >> 5;
    long row = (long)blockIdx.x * 8 + w;             // over B*S rows
    int  b   = (int)(row >> 12);

    float4 sv = ((const float4*)(sf + row * CC))[lane];
    float4 av = ((const float4*)(aggn + (long)b * CC))[lane];
    float4 y  = make_float4(sv.x + av.x, sv.y + av.y, sv.z + av.z, sv.w + av.w);

    float s = y.x + y.y + y.z + y.w;
    for (int m = 16; m; m >>= 1) s += __shfl_xor(s, m, 32);
    float mu = s * (1.0f / 128.0f);

    float dx = y.x - mu, dy = y.y - mu, dz = y.z - mu, dw = y.w - mu;
    float v = dx * dx + dy * dy + dz * dz + dw * dw;
    for (int m = 16; m; m >>= 1) v += __shfl_xor(v, m, 32);
    float rs = rsqrtf(v * (1.0f / 128.0f) + 1e-3f);

    float4 g  = ((const float4*)og)[lane];
    float4 be = ((const float4*)ob)[lane];
    float4 o  = make_float4(dx * rs * g.x + be.x, dy * rs * g.y + be.y,
                            dz * rs * g.z + be.z, dw * rs * g.w + be.w);
    ((float4*)(out + row * CC))[lane] = o;
}

// ---------------------------------------------------------------------------
extern "C" void kernel_launch(void* const* d_in, const int* in_sizes, int n_in,
                              void* d_out, int out_size, void* d_ws, size_t ws_size,
                              hipStream_t stream)
{
    const float* sf   = (const float*)d_in[0];
    const float* qs   = (const float*)d_in[1];
    const float* pe   = (const float*)d_in[2];
    const float* W0   = (const float*)d_in[3];
    const float* b0   = (const float*)d_in[4];
    const float* W1   = (const float*)d_in[5];
    const float* b1   = (const float*)d_in[6];
    const float* Wout = (const float*)d_in[7];
    const float* bout = (const float*)d_in[8];
    const float* mn_g = (const float*)d_in[9];
    const float* mn_b = (const float*)d_in[10];
    const float* on_g = (const float*)d_in[11];
    const float* on_b = (const float*)d_in[12];
    float* out = (float*)d_out;

    // Workspace layout (bytes)
    char* ws = (char*)d_ws;
    __bf16* W0T  = (__bf16*)(ws);                       // 512*256*2 = 262144
    __bf16* W1T  = (__bf16*)(ws + 262144);              // 512*512*2 = 524288
    __bf16* WoT  = (__bf16*)(ws + 786432);              // 128*512*2 = 131072
    float*  agg  = (float*) (ws + 917504);              // 64*128*4  =  32768
    float*  aggn = (float*) (ws + 950272);              // 64*128*4  =  32768

    convert_weights<<<(512 * 512) / 256, 256, 0, stream>>>(
        W0, W1, Wout, W0T, W1T, WoT, agg);

    fused_mlp<<<(BB * SS) / TOK, 256, 0, stream>>>(
        sf, qs, pe, W0T, b0, W1T, b1, WoT, bout, agg);

    agg_ln<<<BB, 128, 0, stream>>>(agg, mn_g, mn_b, aggn);

    out_ln<<<(BB * SS) / 8, 256, 0, stream>>>(sf, aggn, on_g, on_b, out);
}
